// MoETransformerDecoderBlock_13262859010804
// MI455X (gfx1250) — matrix operand order repaired and checked by reading, not compiled, not run on hardware
//
#include <hip/hip_runtime.h>
#include <hip/hip_bf16.h>
#include <math.h>

// ---------------------------------------------------------------------------
// MoE Transformer decoder block for gfx1250 (MI455X), bf16 WMMA + fp32 accum,
// async global->LDS tile staging (ASYNCcnt) for A/Q/K tiles.
// ---------------------------------------------------------------------------

typedef __bf16 v16bf __attribute__((ext_vector_type(16)));
typedef float  v8f   __attribute__((ext_vector_type(8)));
typedef unsigned int u32x4 __attribute__((ext_vector_type(4)));

#define B_ 4
#define S_ 2048
#define D_ 1024
#define H_ 16
#define E_ 8
#define F_ 4096
#define T_ (B_ * S_)
#define CAP_ 2560
#define SLOTS_ (E_ * CAP_)

__device__ __forceinline__ unsigned short f32_to_bf16(float f) {
  unsigned int u = __float_as_uint(f);
  u += 0x7FFFu + ((u >> 16) & 1u);
  return (unsigned short)(u >> 16);
}

union FragAB {
  v16bf v;
  u32x4 u[2];
  unsigned short s[16];
};

__device__ __forceinline__ v8f vzero8() {
  v8f z;
#pragma unroll
  for (int i = 0; i < 8; ++i) z[i] = 0.0f;
  return z;
}

// Async 16B global -> LDS copy (GLOBAL_LOAD_ASYNC_TO_LDS_B128, ASYNCcnt).
// LDS operand: low 32 bits of the generic pointer == AS3 byte offset
// (addrspacecast AS3->flat keeps the offset in the low DWORD).
__device__ __forceinline__ void async_copy_b128(const void* gsrc, void* ldst) {
  asm volatile("global_load_async_to_lds_b128 %0, %1, off"
               :
               : "v"((unsigned)(unsigned long long)(uintptr_t)ldst),
                 "v"((unsigned long long)(uintptr_t)gsrc)
               : "memory");
}
__device__ __forceinline__ void wait_async0() {
  asm volatile("s_wait_asynccnt 0x0" ::: "memory");
}

// ------------------------------- cast --------------------------------------
__global__ void cast_f32_bf16(const float* __restrict__ src,
                              unsigned short* __restrict__ dst, long long n) {
  long long i = (long long)blockIdx.x * blockDim.x + threadIdx.x;
  long long stride = (long long)gridDim.x * blockDim.x;
  for (; i < n; i += stride) dst[i] = f32_to_bf16(src[i]);
}

__global__ void zero_i32(int* __restrict__ p, int n) {
  int i = blockIdx.x * blockDim.x + threadIdx.x;
  if (i < n) p[i] = 0;
}

// ------------------------------- GEMM --------------------------------------
// C[M,N] = epilogue(A[M,K] @ B[K,N] + bias[N]); A/B bf16, WMMA bf16 -> f32.
// blockIdx.z selects an "expert" slice (strides in elements); rowidx gathers
// A rows; cntArr bounds valid rows per slice (tiles past the bound early-out).
template <bool GELU, bool OUT_BF16>
__global__ __launch_bounds__(256) void gemm_bf16(
    const unsigned short* __restrict__ A, long long strideAz,
    const unsigned short* __restrict__ Bm, long long strideBz,
    const float* __restrict__ bias, long long strideBiasZ,
    void* __restrict__ Cout, long long strideCz,
    int M, int N, int K,
    const int* __restrict__ rowidx, int rowidxStrideZ,
    const int* __restrict__ cntArr) {
  const int e = blockIdx.z;
  int bound = M;
  if (cntArr) {
    const int c = cntArr[e];
    bound = (c < M) ? c : M;
  }
  const int m0 = blockIdx.y * 128;
  if (m0 >= bound) return;  // uniform: safe before barriers
  const int n0 = blockIdx.x * 128;

  const unsigned short* Ab = A + (long long)e * strideAz;
  const unsigned short* Bb = Bm + (long long)e * strideBz;
  const float* biasb = bias ? (bias + (long long)e * strideBiasZ) : nullptr;
  const int* ridx = rowidx ? (rowidx + (long long)e * rowidxStrideZ) : nullptr;

  __shared__ unsigned short As[128][40];  // [m][k], +8 pad keeps 16B align
  __shared__ unsigned short Bs[128][40];  // [n][k] (transposed for B frags)

  const int tid = threadIdx.x;
  const int wv = tid >> 5;
  const int lane = tid & 31;
  const int wm = (wv & 1) * 64;
  const int wn = (wv >> 1) * 32;
  const int lr = lane & 15;
  const int kh8 = (lane >> 4) * 8;
  const int kh16 = (lane >> 4) * 16;

  // A tile loader: 128 rows x 32 k, 2 threads per row, 16 bf16 each.
  const int ar = tid >> 1;
  const int ah = (tid & 1) * 16;
  int srcRow = m0 + ar;
  if (srcRow >= bound) srcRow = m0;  // clamp to a valid row (m0 < bound)
  const int tokRow = ridx ? ridx[srcRow] : srcRow;
  const unsigned short* Aptr = Ab + (long long)tokRow * K + ah;

  // B tile loader: 32 k-rows x 128 n, 8 threads per k-row, 16 bf16 each.
  const int bk = tid >> 3;
  const int bn = (tid & 7) * 16;
  const unsigned short* Bptr = Bb + (long long)bk * N + n0 + bn;

  v8f acc[4][2];
#pragma unroll
  for (int i = 0; i < 4; ++i)
#pragma unroll
    for (int j = 0; j < 2; ++j) acc[i][j] = vzero8();

  for (int k0 = 0; k0 < K; k0 += 32) {
    // A tile: async DMA straight into LDS (no VGPR bounce).
    async_copy_b128(Aptr + k0, &As[ar][ah]);
    async_copy_b128(Aptr + k0 + 8, &As[ar][ah + 8]);
    // B tile: VGPR path (needs transpose into [n][k]).
    FragAB bvr;
    bvr.u[0] = *(const u32x4*)(Bptr + (long long)k0 * N);
    bvr.u[1] = *(const u32x4*)(Bptr + (long long)k0 * N + 8);
    if (k0 + 32 < K) {  // gfx1250 global_prefetch of next B K-slab
      __builtin_prefetch(Bptr + (long long)(k0 + 32) * N, 0, 3);
    }
#pragma unroll
    for (int j = 0; j < 16; ++j) Bs[bn + j][bk] = bvr.s[j];
    wait_async0();
    __syncthreads();

    FragAB af[4];
#pragma unroll
    for (int sm = 0; sm < 4; ++sm) {
      const int m = wm + sm * 16 + lr;
      af[sm].u[0] = *(const u32x4*)&As[m][kh8];
      af[sm].u[1] = *(const u32x4*)&As[m][16 + kh8];
    }
    FragAB bf2[2];
#pragma unroll
    for (int sn = 0; sn < 2; ++sn) {
      const int n = wn + sn * 16 + lr;
      bf2[sn].u[0] = *(const u32x4*)&Bs[n][kh8];
      bf2[sn].u[1] = *(const u32x4*)&Bs[n][16 + kh8];
    }
#pragma unroll
    for (int sm = 0; sm < 4; ++sm)
#pragma unroll
      for (int sn = 0; sn < 2; ++sn)
        acc[sm][sn] = __builtin_amdgcn_wmma_f32_16x16x32_bf16(
            false, af[sm].v, false, bf2[sn].v, (short)0, acc[sm][sn], false,
            false);
    __syncthreads();
  }

  // Epilogue: C-fragment layout -> row = (lane/16)*8 + i, col = lane%16.
  const int rh = (lane >> 4) * 8;
#pragma unroll
  for (int sm = 0; sm < 4; ++sm) {
#pragma unroll
    for (int sn = 0; sn < 2; ++sn) {
      const int c = n0 + wn + sn * 16 + lr;
      const float bvv = biasb ? biasb[c] : 0.0f;
#pragma unroll
      for (int i = 0; i < 8; ++i) {
        const int r = m0 + wm + sm * 16 + rh + i;
        if (r < bound) {
          float v = acc[sm][sn][i] + bvv;
          if (GELU) v = 0.5f * v * (1.0f + erff(v * 0.70710678118654752f));
          if (OUT_BF16)
            ((unsigned short*)Cout)[(long long)e * strideCz +
                                    (long long)r * N + c] = f32_to_bf16(v);
          else
            ((float*)Cout)[(long long)e * strideCz + (long long)r * N + c] = v;
        }
      }
    }
  }
}

// --------------------------- flash attention -------------------------------
// One block (4 waves) per (64-row q-tile, head, batch). Each wave owns 16
// q-rows; online softmax in fp32; P re-shaped into A-fragments via LDS.
__global__ __launch_bounds__(128) void flash_attn(
    const unsigned short* __restrict__ Qb, const unsigned short* __restrict__ Kb,
    const unsigned short* __restrict__ Vb, float* __restrict__ Ob) {
  const int qt = blockIdx.x;
  const int h = blockIdx.y;
  const int b = blockIdx.z;
  const int tid = threadIdx.x;
  const int wv = tid >> 5;
  const int lane = tid & 31;
  const int lr = lane & 15;
  const int half = lane >> 4;

  __shared__ unsigned short Qs[64][72];
  __shared__ unsigned short Ks[64][72];
  __shared__ unsigned short Vs[64][72];       // transposed: [dh][krow]
  __shared__ unsigned short Ps[4][16][72];    // per-wave P strip

  {
    const int r = tid >> 1, hf = tid & 1;
    const unsigned short* qp =
        Qb + ((long long)(b * S_ + qt * 64 + r)) * D_ + h * 64 + hf * 32;
#pragma unroll
    for (int j = 0; j < 4; ++j)
      async_copy_b128(qp + j * 8, &Qs[r][hf * 32 + j * 8]);
  }
  wait_async0();
  __syncthreads();

  v8f o[4];
  float mrow[8], lrow[8];
#pragma unroll
  for (int n = 0; n < 4; ++n) o[n] = vzero8();
#pragma unroll
  for (int i = 0; i < 8; ++i) {
    mrow[i] = -1e30f;
    lrow[i] = 0.0f;
  }
  const float scale = 0.125f;  // 1/sqrt(dh=64)

  for (int t0 = 0; t0 < S_; t0 += 64) {
    {
      const int r = tid >> 1, hf = tid & 1;
      const unsigned short* kp =
          Kb + ((long long)(b * S_ + t0 + r)) * D_ + h * 64 + hf * 32;
#pragma unroll
      for (int j = 0; j < 4; ++j)
        async_copy_b128(kp + j * 8, &Ks[r][hf * 32 + j * 8]);
      const unsigned short* vp =
          Vb + ((long long)(b * S_ + t0 + r)) * D_ + h * 64 + hf * 32;
      union {
        u32x4 q[4];
        unsigned short s16[32];
      } vvv;
      const u32x4* s2 = (const u32x4*)vp;
#pragma unroll
      for (int j = 0; j < 4; ++j) vvv.q[j] = s2[j];
#pragma unroll
      for (int j = 0; j < 32; ++j) Vs[hf * 32 + j][r] = vvv.s16[j];
    }
    wait_async0();
    __syncthreads();

    // S = Q K^T : 16 x 64 per wave
    v8f sc[4];
#pragma unroll
    for (int n = 0; n < 4; ++n) sc[n] = vzero8();
#pragma unroll
    for (int kk = 0; kk < 2; ++kk) {
      FragAB af;
      const int m = wv * 16 + lr;
      af.u[0] = *(const u32x4*)&Qs[m][kk * 32 + half * 8];
      af.u[1] = *(const u32x4*)&Qs[m][kk * 32 + 16 + half * 8];
#pragma unroll
      for (int n = 0; n < 4; ++n) {
        FragAB bf;
        const int nn = n * 16 + lr;
        bf.u[0] = *(const u32x4*)&Ks[nn][kk * 32 + half * 8];
        bf.u[1] = *(const u32x4*)&Ks[nn][kk * 32 + 16 + half * 8];
        sc[n] = __builtin_amdgcn_wmma_f32_16x16x32_bf16(
            false, af.v, false, bf.v, (short)0, sc[n], false, false);
      }
    }

    // online softmax (row stats live redundantly on the 16 lanes of a half)
#pragma unroll
    for (int n = 0; n < 4; ++n) sc[n] *= scale;
#pragma unroll
    for (int i = 0; i < 8; ++i) {
      float tm = fmaxf(fmaxf(sc[0][i], sc[1][i]), fmaxf(sc[2][i], sc[3][i]));
#pragma unroll
      for (int msk = 1; msk < 16; msk <<= 1)
        tm = fmaxf(tm, __shfl_xor(tm, msk, 32));
      const float mn = fmaxf(mrow[i], tm);
      const float corr = __expf(mrow[i] - mn);
      mrow[i] = mn;
      float rs = 0.0f;
#pragma unroll
      for (int n = 0; n < 4; ++n) {
        const float p = __expf(sc[n][i] - mn);
        sc[n][i] = p;
        rs += p;
      }
#pragma unroll
      for (int msk = 1; msk < 16; msk <<= 1) rs += __shfl_xor(rs, msk, 32);
      lrow[i] = lrow[i] * corr + rs;
#pragma unroll
      for (int n = 0; n < 4; ++n) o[n][i] *= corr;
#pragma unroll
      for (int n = 0; n < 4; ++n)
        Ps[wv][half * 8 + i][n * 16 + lr] = f32_to_bf16(sc[n][i]);
    }

    // O += P V  (per-wave LDS strip; DS ops of a wave are in-order)
#pragma unroll
    for (int kk = 0; kk < 2; ++kk) {
      FragAB af;
      af.u[0] = *(const u32x4*)&Ps[wv][lr][kk * 32 + half * 8];
      af.u[1] = *(const u32x4*)&Ps[wv][lr][kk * 32 + 16 + half * 8];
#pragma unroll
      for (int n = 0; n < 4; ++n) {
        FragAB bf;
        const int nn = n * 16 + lr;
        bf.u[0] = *(const u32x4*)&Vs[nn][kk * 32 + half * 8];
        bf.u[1] = *(const u32x4*)&Vs[nn][kk * 32 + 16 + half * 8];
        o[n] = __builtin_amdgcn_wmma_f32_16x16x32_bf16(
            false, af.v, false, bf.v, (short)0, o[n], false, false);
      }
    }
    __syncthreads();
  }

#pragma unroll
  for (int i = 0; i < 8; ++i) {
    const float inv = 1.0f / lrow[i];
    const int row = qt * 64 + wv * 16 + half * 8 + i;
#pragma unroll
    for (int n = 0; n < 4; ++n)
      Ob[((long long)(b * S_ + row)) * D_ + h * 64 + n * 16 + lr] =
          o[n][i] * inv;
  }
}

// ------------------------ residual + layernorm -----------------------------
__global__ __launch_bounds__(256) void residual_ln(
    const float* __restrict__ xin, const float* __restrict__ y,
    const float* __restrict__ g, const float* __restrict__ bta,
    float* __restrict__ xout, unsigned short* __restrict__ xbf) {
  const int t = blockIdx.x;
  const int tid = threadIdx.x;
  __shared__ float red[256];
  const float* yr = y + (long long)t * D_;
  float v[4];
  float s = 0.0f;
#pragma unroll
  for (int j = 0; j < 4; ++j) {
    v[j] = yr[tid + j * 256];
    s += v[j];
  }
  red[tid] = s;
  __syncthreads();
  for (int st = 128; st > 0; st >>= 1) {
    if (tid < st) red[tid] += red[tid + st];
    __syncthreads();
  }
  const float mu = red[0] * (1.0f / D_);
  __syncthreads();
  float ss = 0.0f;
#pragma unroll
  for (int j = 0; j < 4; ++j) {
    const float d = v[j] - mu;
    ss += d * d;
  }
  red[tid] = ss;
  __syncthreads();
  for (int st = 128; st > 0; st >>= 1) {
    if (tid < st) red[tid] += red[tid + st];
    __syncthreads();
  }
  const float rinv = rsqrtf(red[0] * (1.0f / D_) + 1e-5f);
#pragma unroll
  for (int j = 0; j < 4; ++j) {
    const int d = tid + j * 256;
    const float val =
        xin[(long long)t * D_ + d] + g[d] * ((v[j] - mu) * rinv) + bta[d];
    xout[(long long)t * D_ + d] = val;
    if (xbf) xbf[(long long)t * D_ + d] = f32_to_bf16(val);
  }
}

// ------------------------------- gating ------------------------------------
__global__ __launch_bounds__(128) void gate_topk(
    const float* __restrict__ x1, const float* __restrict__ gw,
    const float* __restrict__ gb, int* __restrict__ cnt,
    int* __restrict__ perm, int* __restrict__ slot_of,
    float* __restrict__ wts) {
  const int t = blockIdx.x;
  const int tid = threadIdx.x;
  __shared__ float red[128][8];
  float acc[8];
#pragma unroll
  for (int e = 0; e < 8; ++e) acc[e] = 0.0f;
  const float* xr = x1 + (long long)t * D_;
  for (int d = tid; d < D_; d += 128) {
    const float xv = xr[d];
    const float* g = gw + (long long)d * E_;
#pragma unroll
    for (int e = 0; e < 8; ++e) acc[e] += xv * g[e];
  }
#pragma unroll
  for (int e = 0; e < 8; ++e) red[tid][e] = acc[e];
  __syncthreads();
  for (int st = 64; st > 0; st >>= 1) {
    if (tid < st) {
#pragma unroll
      for (int e = 0; e < 8; ++e) red[tid][e] += red[tid + st][e];
    }
    __syncthreads();
  }
  if (tid == 0) {
    float l[8];
#pragma unroll
    for (int e = 0; e < 8; ++e) l[e] = red[0][e] + gb[e];
    float mx = l[0];
#pragma unroll
    for (int e = 1; e < 8; ++e) mx = fmaxf(mx, l[e]);
    float p[8];
    float sum = 0.0f;
#pragma unroll
    for (int e = 0; e < 8; ++e) {
      p[e] = __expf(l[e] - mx);
      sum += p[e];
    }
    const float inv = 1.0f / sum;
    int i0 = 0;
    float v0 = p[0];
#pragma unroll
    for (int e = 1; e < 8; ++e)
      if (p[e] > v0) {
        v0 = p[e];
        i0 = e;
      }
    int i1 = (i0 == 0) ? 1 : 0;
    float v1 = p[i1];
#pragma unroll
    for (int e = 0; e < 8; ++e)
      if (e != i0 && p[e] > v1) {
        v1 = p[e];
        i1 = e;
      }
    v0 *= inv;
    v1 *= inv;
    const int p0 = atomicAdd(&cnt[i0], 1);
    if (p0 < CAP_) {
      perm[i0 * CAP_ + p0] = t;
      slot_of[2 * t] = i0 * CAP_ + p0;
    } else {
      slot_of[2 * t] = -1;
    }
    wts[2 * t] = v0;
    const int p1 = atomicAdd(&cnt[i1], 1);
    if (p1 < CAP_) {
      perm[i1 * CAP_ + p1] = t;
      slot_of[2 * t + 1] = i1 * CAP_ + p1;
    } else {
      slot_of[2 * t + 1] = -1;
    }
    wts[2 * t + 1] = v1;
  }
}

// --------------------- combine + layernorm2 + residual ---------------------
__global__ __launch_bounds__(256) void moe_combine_ln(
    const float* __restrict__ x1, const float* __restrict__ yslot,
    const int* __restrict__ slot_of, const float* __restrict__ wts,
    const float* __restrict__ g, const float* __restrict__ bta,
    float* __restrict__ out) {
  const int t = blockIdx.x;
  const int tid = threadIdx.x;
  __shared__ float red[256];
  const int s0 = slot_of[2 * t], s1 = slot_of[2 * t + 1];
  const float w0 = wts[2 * t], w1 = wts[2 * t + 1];
  float v[4];
  float s = 0.0f;
#pragma unroll
  for (int j = 0; j < 4; ++j) {
    const int d = tid + j * 256;
    float m = 0.0f;
    if (s0 >= 0) m += w0 * yslot[(long long)s0 * D_ + d];
    if (s1 >= 0) m += w1 * yslot[(long long)s1 * D_ + d];
    v[j] = m;
    s += m;
  }
  red[tid] = s;
  __syncthreads();
  for (int st = 128; st > 0; st >>= 1) {
    if (tid < st) red[tid] += red[tid + st];
    __syncthreads();
  }
  const float mu = red[0] * (1.0f / D_);
  __syncthreads();
  float ss = 0.0f;
#pragma unroll
  for (int j = 0; j < 4; ++j) {
    const float d = v[j] - mu;
    ss += d * d;
  }
  red[tid] = ss;
  __syncthreads();
  for (int st = 128; st > 0; st >>= 1) {
    if (tid < st) red[tid] += red[tid + st];
    __syncthreads();
  }
  const float rinv = rsqrtf(red[0] * (1.0f / D_) + 1e-5f);
#pragma unroll
  for (int j = 0; j < 4; ++j) {
    const int d = tid + j * 256;
    out[(long long)t * D_ + d] =
        x1[(long long)t * D_ + d] + g[d] * ((v[j] - mu) * rinv) + bta[d];
  }
}

// ------------------------------ launcher -----------------------------------
extern "C" void kernel_launch(void* const* d_in, const int* in_sizes, int n_in,
                              void* d_out, int out_size, void* d_ws,
                              size_t ws_size, hipStream_t stream) {
  (void)in_sizes;
  (void)n_in;
  (void)out_size;
  (void)ws_size;
  const float* x = (const float*)d_in[0];
  const float* Wq = (const float*)d_in[1];
  const float* Wk = (const float*)d_in[2];
  const float* Wv = (const float*)d_in[3];
  const float* Wo = (const float*)d_in[4];
  const float* bq = (const float*)d_in[5];
  const float* bk = (const float*)d_in[6];
  const float* bv = (const float*)d_in[7];
  const float* bo = (const float*)d_in[8];
  const float* ln1g = (const float*)d_in[9];
  const float* ln1b = (const float*)d_in[10];
  const float* ln2g = (const float*)d_in[11];
  const float* ln2b = (const float*)d_in[12];
  const float* gw = (const float*)d_in[13];
  const float* gb = (const float*)d_in[14];
  const float* W1 = (const float*)d_in[15];
  const float* b1 = (const float*)d_in[16];
  const float* W2 = (const float*)d_in[17];
  const float* b2 = (const float*)d_in[18];
  float* out = (float*)d_out;

  char* base = (char*)d_ws;
  size_t off = 0;
  auto alloc = [&](size_t bytes) -> void* {
    void* p = base + off;
    off = (off + bytes + 255) & ~(size_t)255;
    return p;
  };

  unsigned short* xbf = (unsigned short*)alloc((size_t)T_ * D_ * 2);
  unsigned short* wqbf = (unsigned short*)alloc((size_t)D_ * D_ * 2);
  unsigned short* wkbf = (unsigned short*)alloc((size_t)D_ * D_ * 2);
  unsigned short* wvbf = (unsigned short*)alloc((size_t)D_ * D_ * 2);
  unsigned short* wobf = (unsigned short*)alloc((size_t)D_ * D_ * 2);
  unsigned short* qbf = (unsigned short*)alloc((size_t)T_ * D_ * 2);
  unsigned short* kbf = (unsigned short*)alloc((size_t)T_ * D_ * 2);
  unsigned short* vbf = (unsigned short*)alloc((size_t)T_ * D_ * 2);
  float* attnf = (float*)alloc((size_t)T_ * D_ * 4);
  unsigned short* attnbf = (unsigned short*)alloc((size_t)T_ * D_ * 2);
  float* projf = (float*)alloc((size_t)T_ * D_ * 4);
  float* x1f = (float*)alloc((size_t)T_ * D_ * 4);
  unsigned short* x1bf = (unsigned short*)alloc((size_t)T_ * D_ * 2);
  unsigned short* w1bf = (unsigned short*)alloc((size_t)E_ * D_ * F_ * 2);
  unsigned short* w2bf = (unsigned short*)alloc((size_t)E_ * F_ * D_ * 2);
  unsigned short* h1bf = (unsigned short*)alloc((size_t)SLOTS_ * F_ * 2);
  float* yslot = (float*)alloc((size_t)SLOTS_ * D_ * 4);
  int* cnt = (int*)alloc((size_t)E_ * 4);
  int* perm = (int*)alloc((size_t)SLOTS_ * 4);
  int* slot_of = (int*)alloc((size_t)T_ * 2 * 4);
  float* wts = (float*)alloc((size_t)T_ * 2 * 4);

  const int CB = 256;
  const int CG = 2048;

  // 1) casts of x + weights to bf16
  cast_f32_bf16<<<CG, CB, 0, stream>>>(x, xbf, (long long)T_ * D_);
  cast_f32_bf16<<<CG, CB, 0, stream>>>(Wq, wqbf, (long long)D_ * D_);
  cast_f32_bf16<<<CG, CB, 0, stream>>>(Wk, wkbf, (long long)D_ * D_);
  cast_f32_bf16<<<CG, CB, 0, stream>>>(Wv, wvbf, (long long)D_ * D_);
  cast_f32_bf16<<<CG, CB, 0, stream>>>(Wo, wobf, (long long)D_ * D_);
  cast_f32_bf16<<<CG, CB, 0, stream>>>(W1, w1bf, (long long)E_ * D_ * F_);
  cast_f32_bf16<<<CG, CB, 0, stream>>>(W2, w2bf, (long long)E_ * F_ * D_);

  // 2) Q/K/V projections (bf16 outputs)
  dim3 gq(D_ / 128, T_ / 128, 1);
  gemm_bf16<false, true><<<gq, 256, 0, stream>>>(
      xbf, 0, wqbf, 0, bq, 0, (void*)qbf, 0, T_, D_, D_, nullptr, 0, nullptr);
  gemm_bf16<false, true><<<gq, 256, 0, stream>>>(
      xbf, 0, wkbf, 0, bk, 0, (void*)kbf, 0, T_, D_, D_, nullptr, 0, nullptr);
  gemm_bf16<false, true><<<gq, 256, 0, stream>>>(
      xbf, 0, wvbf, 0, bv, 0, (void*)vbf, 0, T_, D_, D_, nullptr, 0, nullptr);

  // 3) flash attention
  flash_attn<<<dim3(S_ / 64, H_, B_), 128, 0, stream>>>(qbf, kbf, vbf, attnf);

  // 4) output projection + LN1 + residual
  cast_f32_bf16<<<CG, CB, 0, stream>>>(attnf, attnbf, (long long)T_ * D_);
  gemm_bf16<false, false><<<gq, 256, 0, stream>>>(
      attnbf, 0, wobf, 0, bo, 0, (void*)projf, 0, T_, D_, D_, nullptr, 0,
      nullptr);
  residual_ln<<<T_, 256, 0, stream>>>(x, projf, ln1g, ln1b, x1f, x1bf);

  // 5) gating + top-2 routing
  zero_i32<<<1, 32, 0, stream>>>(cnt, E_);
  gate_topk<<<T_, 128, 0, stream>>>(x1f, gw, gb, cnt, perm, slot_of, wts);

  // 6) expert GEMM1 (gathered rows, GELU, bf16 out)
  gemm_bf16<true, true><<<dim3(F_ / 128, CAP_ / 128, E_), 256, 0, stream>>>(
      x1bf, 0, w1bf, (long long)D_ * F_, b1, F_, (void*)h1bf,
      (long long)CAP_ * F_, CAP_, F_, D_, perm, CAP_, cnt);

  // 7) expert GEMM2 (slot-dense rows, fp32 out)
  gemm_bf16<false, false><<<dim3(D_ / 128, CAP_ / 128, E_), 256, 0, stream>>>(
      h1bf, (long long)CAP_ * F_, w2bf, (long long)F_ * D_, b2, D_,
      (void*)yslot, (long long)CAP_ * D_, CAP_, D_, F_, nullptr, 0, cnt);

  // 8) combine + LN2 + residual -> output
  moe_combine_ln<<<T_, 256, 0, stream>>>(x1f, yslot, slot_of, wts, ln2g, ln2b,
                                         out);
}
